// SparseVSCBlockRulebook_50354196578891
// MI455X (gfx1250) — matrix operand order repaired and checked by reading, not compiled, not run on hardware
//
#include <hip/hip_runtime.h>

typedef __attribute__((ext_vector_type(16))) __bf16        v16bf;
typedef __attribute__((ext_vector_type(8)))  float         v8f;
typedef __attribute__((ext_vector_type(8)))  unsigned int  v8u;

#define CCH   64      // C_IN == C_OUT == 64
#define KTAPS 9

// pack two f32 -> two bf16 (RNE) in one dword: round, then v_perm high halves
static __device__ inline unsigned pkbf(float a, float b) {
    unsigned ua = __builtin_bit_cast(unsigned, a);
    unsigned ub = __builtin_bit_cast(unsigned, b);
    ua = ua + 0x7FFFu + ((ua >> 16) & 1u);
    ub = ub + 0x7FFFu + ((ub >> 16) & 1u);
    // bytes: [ub3 ub2 ua3 ua2]  ->  {bf16(b), bf16(a)}
    return __builtin_amdgcn_perm(ub, ua, 0x07060302u);
}

static __device__ inline __bf16 f2bf(float f) {
    unsigned u = __builtin_bit_cast(unsigned, f);
    u = u + 0x7FFFu + ((u >> 16) & 1u);
    unsigned short h = (unsigned short)(u >> 16);
    return __builtin_bit_cast(__bf16, h);
}

// ---------------- kernel 0: zero accumulator + stats, copy coords ----------
__global__ void vsc_init(const int* __restrict__ coords, int* __restrict__ coords_out,
                         float* __restrict__ outbuf, float* __restrict__ stats, int N) {
    long long stride = (long long)gridDim.x * blockDim.x;
    long long gid = (long long)blockIdx.x * blockDim.x + threadIdx.x;
    long long n64 = (long long)N * CCH;
    for (long long i = gid; i < n64; i += stride) outbuf[i] = 0.0f;
    long long n3 = (long long)N * 3;
    for (long long i = gid; i < n3; i += stride) coords_out[i] = coords[i];
    if (gid < 128) stats[gid] = 0.0f;
}

// ---------------- A fragment: 16x32 bf16, per ISA layout -------------------
// lane L (row M = L&15): k0 = (L>>4)*8 (+chunk*32); elems 0..7 = K k0..k0+7,
// elems 8..15 = K k0+16..k0+23  -> two contiguous 32B reads, packed as dwords.
static __device__ inline v16bf load_a_frag(const float* __restrict__ fb, int k0) {
    float4 q0 = *(const float4*)(fb + k0);
    float4 q1 = *(const float4*)(fb + k0 + 4);
    float4 q2 = *(const float4*)(fb + k0 + 16);
    float4 q3 = *(const float4*)(fb + k0 + 20);
    v8u a;
    a[0] = pkbf(q0.x, q0.y); a[1] = pkbf(q0.z, q0.w);
    a[2] = pkbf(q1.x, q1.y); a[3] = pkbf(q1.z, q1.w);
    a[4] = pkbf(q2.x, q2.y); a[5] = pkbf(q2.z, q2.w);
    a[6] = pkbf(q3.x, q3.y); a[7] = pkbf(q3.z, q3.w);
    return __builtin_bit_cast(v16bf, a);
}

// ---------------- kernel 1: gather -> WMMA GEMM -> scatter-add -------------
// Full 16-row tiles ONLY: no bounds checks anywhere, so the scatter is 32
// back-to-back global_atomic_add_f32 with no exec-mask games.
// LDS holds weight[k] pre-swizzled into B-fragment order:
//   frag = c*4 + n ; offset(frag, lane, e) = frag*512 + lane*16 + e (bf16)
// -> each lane's 16 B elements are 32 contiguous bytes = 2x ds_load_b128.
__global__ void __launch_bounds__(256)
vsc_gemm_scatter(const float* __restrict__ feats, const int* __restrict__ in_rows,
                 const int* __restrict__ out_rows, const float* __restrict__ weight,
                 float* __restrict__ outbuf, int M, int MTfull) {
    __shared__ __bf16 wlds[8 * 512];             // 8 KB
    int k = blockIdx.y;

    const float* wk = weight + (size_t)k * CCH * CCH;
    for (int j = threadIdx.x; j < CCH * CCH; j += blockDim.x) {
        int i = j >> 6, o = j & 63;              // weight[k][i][o], coalesced read
        int c    = i >> 5;
        int r5   = i & 31;
        int lane = (((r5 >> 3) & 1) << 4) | (o & 15);
        int e    = (r5 & 7) | ((r5 & 16) >> 1);
        int frag = c * 4 + (o >> 4);
        wlds[frag * 512 + lane * 16 + e] = f2bf(wk[j]);
    }
    __syncthreads();

    int wave = threadIdx.x >> 5;
    int lane = threadIdx.x & 31;
    int mtile = blockIdx.x * 8 + wave;
    if (mtile >= MTfull) return;                 // wave-uniform: EXEC all-1 below
    int mbase = mtile * 16;
    int col = lane & 15;
    int hi  = lane >> 4;

    // 8 out-row broadcasts as two b128 loads, then 8 base addresses
    const int* outr = out_rows + (size_t)k * M + mbase + hi * 8;
    int4 oa = *(const int4*)(outr);
    int4 ob = *(const int4*)(outr + 4);
    int orows[8] = {oa.x, oa.y, oa.z, oa.w, ob.x, ob.y, ob.z, ob.w};
    float* obase[8];
#pragma unroll
    for (int r = 0; r < 8; ++r)
        obase[r] = outbuf + (size_t)orows[r] * CCH + col;

    // gather A rows (random rows of feats, 256B each)
    const int* inr = in_rows + (size_t)k * M + mbase;
    int arow = inr[col];
    const float* fbase = feats + (size_t)arow * CCH;
    v16bf a0 = load_a_frag(fbase, hi * 8);       // K 0..31
    v16bf a1 = load_a_frag(fbase, 32 + hi * 8);  // K 32..63

#pragma unroll
    for (int n = 0; n < 4; ++n) {
        v16bf b0 = __builtin_bit_cast(v16bf, *(const v8u*)(wlds + (0 + n) * 512 + lane * 16));
        v16bf b1 = __builtin_bit_cast(v16bf, *(const v8u*)(wlds + (4 + n) * 512 + lane * 16));
        v8f acc = {};
        acc = __builtin_amdgcn_wmma_f32_16x16x32_bf16(false, a0, false, b0,
                                                      (short)0, acc, false, false);
        acc = __builtin_amdgcn_wmma_f32_16x16x32_bf16(false, a1, false, b1,
                                                      (short)0, acc, false, false);
#pragma unroll
        for (int r = 0; r < 8; ++r)              // D VGPR r -> tile row r + hi*8
            unsafeAtomicAdd(obase[r] + n * 16, acc[r]);
    }
}

// ---------------- kernel 1b: scalar tail for M % 16 rows (rarely launched) -
__global__ void vsc_tail(const float* __restrict__ feats, const int* __restrict__ in_rows,
                         const int* __restrict__ out_rows, const float* __restrict__ weight,
                         float* __restrict__ outbuf, int M, int m0) {
    int rem = M - m0;                            // < 16
    int total = rem * KTAPS * CCH;
    for (int idx = blockIdx.x * blockDim.x + threadIdx.x; idx < total;
         idx += gridDim.x * blockDim.x) {
        int o = idx & 63;
        int j = idx >> 6;
        int k = j / rem;
        int m = m0 + (j - k * rem);
        int ar  = in_rows[(size_t)k * M + m];
        int orw = out_rows[(size_t)k * M + m];
        const float* f = feats + (size_t)ar * CCH;
        const float* w = weight + (size_t)k * CCH * CCH + o;
        float s = 0.0f;
        for (int i = 0; i < CCH; ++i) s += f[i] * w[i * CCH];
        unsafeAtomicAdd(&outbuf[(size_t)orw * CCH + o], s);
    }
}

// ---------------- kernel 2: per-channel sum / sumsq (float4) ---------------
__global__ void vsc_bn_stats_v4(const float* __restrict__ outbuf,
                                float* __restrict__ stats, int N) {
    __shared__ float s1[256 * 4], s2[256 * 4];
    int q = threadIdx.x & 15;                    // channel quad: channels 4q..4q+3
    int g = threadIdx.x >> 4;                    // 16 row-groups per block
    float sx = 0, sy = 0, sz = 0, sw = 0;
    float px = 0, py = 0, pz = 0, pw = 0;
    for (long long row = (long long)blockIdx.x * 16 + g; row < N;
         row += (long long)gridDim.x * 16) {
        float4 x = *(const float4*)(outbuf + row * CCH + q * 4);
        sx += x.x; sy += x.y; sz += x.z; sw += x.w;
        px += x.x * x.x; py += x.y * x.y; pz += x.z * x.z; pw += x.w * x.w;
    }
    int t4 = threadIdx.x * 4;
    s1[t4] = sx; s1[t4 + 1] = sy; s1[t4 + 2] = sz; s1[t4 + 3] = sw;
    s2[t4] = px; s2[t4 + 1] = py; s2[t4 + 2] = pz; s2[t4 + 3] = pw;
    __syncthreads();
    if (threadIdx.x < 64) {                      // one thread per channel
        int qq = threadIdx.x >> 2, cc = threadIdx.x & 3;
        float s = 0, p = 0;
#pragma unroll
        for (int j = 0; j < 16; ++j) {
            int idx = (j * 16 + qq) * 4 + cc;
            s += s1[idx]; p += s2[idx];
        }
        unsafeAtomicAdd(&stats[threadIdx.x], s);
        unsafeAtomicAdd(&stats[64 + threadIdx.x], p);
    }
}

__global__ void vsc_bn_stats(const float* __restrict__ outbuf,
                             float* __restrict__ stats, int N) {
    __shared__ float ssum[256], ssq[256];
    int c = threadIdx.x & 63;
    int g = threadIdx.x >> 6;
    float s = 0.0f, q = 0.0f;
    for (long long row = (long long)blockIdx.x * 4 + g; row < N;
         row += (long long)gridDim.x * 4) {
        float x = outbuf[row * CCH + c];
        s += x; q += x * x;
    }
    ssum[threadIdx.x] = s; ssq[threadIdx.x] = q;
    __syncthreads();
    if (threadIdx.x < 64) {
        s = ssum[threadIdx.x] + ssum[threadIdx.x + 64] +
            ssum[threadIdx.x + 128] + ssum[threadIdx.x + 192];
        q = ssq[threadIdx.x] + ssq[threadIdx.x + 64] +
            ssq[threadIdx.x + 128] + ssq[threadIdx.x + 192];
        unsafeAtomicAdd(&stats[c], s);
        unsafeAtomicAdd(&stats[64 + c], q);
    }
}

// ---------------- kernel 3: BN normalize + ReLU in place -------------------
__global__ void vsc_bn_finalize_v4(float* __restrict__ outbuf, const float* __restrict__ stats,
                                   const float* __restrict__ gamma, const float* __restrict__ beta,
                                   int N) {
    long long total = (long long)N * (CCH / 4);
    long long stride = (long long)gridDim.x * blockDim.x;
    float invN = 1.0f / (float)N;
    float4* ob = (float4*)outbuf;
    for (long long i = (long long)blockIdx.x * blockDim.x + threadIdx.x;
         i < total; i += stride) {
        int c = ((int)(i & 15)) * 4;
        float4 x = ob[i];
        float r[4] = {x.x, x.y, x.z, x.w};
#pragma unroll
        for (int j = 0; j < 4; ++j) {
            float mean = stats[c + j] * invN;    // bias cancels inside BN
            float var  = stats[64 + c + j] * invN - mean * mean;
            float inv  = rsqrtf(var + 1e-5f);
            r[j] = fmaxf((r[j] - mean) * inv * gamma[c + j] + beta[c + j], 0.0f);
        }
        float4 y = {r[0], r[1], r[2], r[3]};
        ob[i] = y;
    }
}

__global__ void vsc_bn_finalize(float* __restrict__ outbuf, const float* __restrict__ stats,
                                const float* __restrict__ gamma, const float* __restrict__ beta,
                                int N) {
    long long total = (long long)N * CCH;
    long long stride = (long long)gridDim.x * blockDim.x;
    float invN = 1.0f / (float)N;
    for (long long i = (long long)blockIdx.x * blockDim.x + threadIdx.x;
         i < total; i += stride) {
        int c = (int)(i & 63);
        float mean = stats[c] * invN;
        float var  = stats[64 + c] * invN - mean * mean;
        float inv  = rsqrtf(var + 1e-5f);
        float y = (outbuf[i] - mean) * inv * gamma[c] + beta[c];
        outbuf[i] = fmaxf(y, 0.0f);
    }
}

extern "C" void kernel_launch(void* const* d_in, const int* in_sizes, int n_in,
                              void* d_out, int out_size, void* d_ws, size_t ws_size,
                              hipStream_t stream) {
    const int*   coords   = (const int*)d_in[0];
    const float* feats    = (const float*)d_in[1];
    const int*   in_rows  = (const int*)d_in[2];
    const int*   out_rows = (const int*)d_in[3];
    const float* weight   = (const float*)d_in[4];
    // d_in[5] = bias (cancels inside BN), d_in[6] = gamma, d_in[7] = beta
    const float* gamma    = (const float*)d_in[6];
    const float* beta     = (const float*)d_in[7];

    int N = in_sizes[1] / CCH;
    int M = in_sizes[2] / KTAPS;
    int MTfull = M / 16;                         // full 16-row tiles
    int m0 = MTfull * 16;                        // first tail row (== M if none)

    float* outf   = (float*)d_out;
    int*   outc   = (int*)d_out;                 // coords bits, first N*3 elems
    float* outbuf = outf + (size_t)N * 3;        // (N, 64) result region
    float* stats  = (float*)d_ws;                // 128 floats: sum | sumsq
    bool aligned16 = (((uintptr_t)outbuf) & 15) == 0;

    vsc_init<<<1024, 256, 0, stream>>>(coords, outc, outbuf, stats, N);

    if (MTfull > 0) {
        dim3 g1((MTfull + 7) / 8, KTAPS);
        vsc_gemm_scatter<<<g1, 256, 0, stream>>>(feats, in_rows, out_rows, weight,
                                                 outbuf, M, MTfull);
    }
    if (m0 < M) {
        vsc_tail<<<36, 256, 0, stream>>>(feats, in_rows, out_rows, weight,
                                         outbuf, M, m0);
    }

    if (aligned16) {
        vsc_bn_stats_v4<<<768, 256, 0, stream>>>(outbuf, stats, N);
        vsc_bn_finalize_v4<<<1024, 256, 0, stream>>>(outbuf, stats, gamma, beta, N);
    } else {
        vsc_bn_stats<<<768, 256, 0, stream>>>(outbuf, stats, N);
        vsc_bn_finalize<<<1024, 256, 0, stream>>>(outbuf, stats, gamma, beta, N);
    }
}
